// GraphicalLipschitzResidualBlock_39092792328828
// MI455X (gfx1250) — matrix operand order repaired and checked by reading, not compiled, not run on hardware
//
#include <hip/hip_runtime.h>
#include <hip/hip_bf16.h>
#include <math.h>

// GraphicalLipschitzResidualBlock for MI455X (gfx1250, wave32, WMMA).
// ~69 GFLOP vs ~5.5 MB traffic -> compute bound -> f16 WMMA, f32 accumulate.
// Stage-2 (per-sample 256x256 @ 256x64 Jacobian GEMM) dominates; W1 lives in
// LDS as f16 and is reused for all 16 samples of the block; two samples per pass halve the
// A-fragment LDS traffic per WMMA.

typedef _Float16 v16h __attribute__((ext_vector_type(16)));
typedef _Float16 v8h  __attribute__((ext_vector_type(8)));
typedef float    v8f  __attribute__((ext_vector_type(8)));

#define NSAMP 8192
#define INDIM 64
#define CONDD 32
#define INSZ  96
#define H1D   256
#define H2D   256

// f16 weight layout inside d_ws (element offsets)
#define WS_W1   0         // 256*256  [i][h]
#define WS_W0   65536     // 256*96   [h][in]
#define WS_W2   90112     // 64*256   [o][h2]
#define WS_W0DT 106496    // 64*256   [o][h1] = W0[h1][o]
#define WS_TOT  122880

__device__ __forceinline__ v16h cat16(v8h lo, v8h hi) {
  return __builtin_shufflevector(lo, hi, 0,1,2,3,4,5,6,7,8,9,10,11,12,13,14,15);
}

// A fragment (16x32 f16, M x K, row-major K-contiguous source).
__device__ __forceinline__ v16h frag_a(const _Float16* base, int ldk, int row,
                                       int k0, int lane) {
  int koff = (lane & 16) ? 8 : 0;
  const _Float16* p = base + row * ldk + k0 + koff;
  v8h lo = *(const v8h*)p;
  v8h hi = *(const v8h*)(p + 16);
  return cat16(lo, hi);
}

// B fragment (32x16 f16, K x N, source stored N-major so K is contiguous).
__device__ __forceinline__ v16h frag_b(const _Float16* baseT, int ldk, int n,
                                       int k0, int lane) {
  int koff = (lane & 16) ? 8 : 0;
  const _Float16* p = baseT + n * ldk + k0 + koff;
  v8h lo = *(const v8h*)p;
  v8h hi = *(const v8h*)(p + 16);
  return cat16(lo, hi);
}

__device__ __forceinline__ v8f wmma16(v16h a, v16h b, v8f c) {
  return __builtin_amdgcn_wmma_f32_16x16x32_f16(false, a, false, b,
                                                (short)0, c, false, false);
}

__device__ __forceinline__ void lipswish_pair(float xv, float beta,
                                              float& a, float& hp) {
  const float inv11 = 1.0f / 1.1f;
  float e  = __expf(-beta * xv);
  float sg = __builtin_amdgcn_rcpf(1.0f + e);   // fast v_rcp_f32
  float f  = xv * sg * inv11;
  a  = f;
  hp = beta * f + sg * (inv11 - beta * f);
}

// Bulk LDS staging via gfx1250 async DMA loads (ASYNCcnt-tracked).
__device__ __forceinline__ void async_copy_lds(_Float16* dst,
                                               const _Float16* src,
                                               int nhalves, int tid) {
  int nchunk = nhalves >> 3;                    // 16-byte chunks
  for (int c = tid; c < nchunk; c += 256) {
    unsigned int ldsoff = (unsigned int)(uintptr_t)(dst + c * 8);
    const void* g = (const void*)(src + c * 8);
    asm volatile("global_load_async_to_lds_b128 %0, %1, off"
                 :: "v"(ldsoff), "v"(g) : "memory");
  }
}

// One-time f32 -> f16 weight conversion (both W0 layouts) into workspace.
__global__ void glrb_convert(const float* __restrict__ W0,
                             const float* __restrict__ W1,
                             const float* __restrict__ W2,
                             _Float16* __restrict__ ws) {
  int i = blockIdx.x * blockDim.x + threadIdx.x;
  int stride = gridDim.x * blockDim.x;
  for (int idx = i; idx < H2D * H1D; idx += stride)
    ws[WS_W1 + idx] = (_Float16)W1[idx];
  for (int idx = i; idx < H1D * INSZ; idx += stride)
    ws[WS_W0 + idx] = (_Float16)W0[idx];
  for (int idx = i; idx < INDIM * H2D; idx += stride)
    ws[WS_W2 + idx] = (_Float16)W2[idx];
  for (int idx = i; idx < INDIM * H1D; idx += stride) {
    int o = idx >> 8, h = idx & 255;
    ws[WS_W0DT + idx] = (_Float16)W0[h * INSZ + o];   // W0d = W0[:, :64]
  }
}

__global__ __launch_bounds__(256, 1)
void glrb_kernel(const float* __restrict__ eps, const float* __restrict__ xc,
                 const _Float16* __restrict__ ws,
                 const float* __restrict__ b0, const float* __restrict__ b1,
                 const float* __restrict__ b2, const float* __restrict__ betap,
                 float* __restrict__ out) {
  __shared__ _Float16 W1h[H2D * H1D];     // 128 KB
  __shared__ _Float16 W0h[H1D * INSZ];    //  48 KB
  __shared__ _Float16 W2h[INDIM * H2D];   //  32 KB
  __shared__ _Float16 W0dTh[INDIM * H1D]; //  32 KB
  __shared__ _Float16 inph[16 * INSZ];    //   3 KB
  __shared__ _Float16 a0h[16 * H1D];      //   8 KB
  __shared__ _Float16 hp0h[16 * H1D];     //   8 KB
  __shared__ _Float16 a1h[16 * H2D];      //   8 KB
  __shared__ _Float16 hp1h[16 * H2D];     //   8 KB
  __shared__ float    ldsDiag[16 * INDIM];//   4 KB

  const int tid  = threadIdx.x;
  const int lane = tid & 31;
  const int wave = tid >> 5;
  const int s0   = blockIdx.x * 16;       // 16 samples per block
  const float beta = betap[0];

  // ---- stage LDS: async DMA for pre-converted f16 weights ----
  async_copy_lds(W1h,   ws + WS_W1,   H2D * H1D,   tid);
  async_copy_lds(W0h,   ws + WS_W0,   H1D * INSZ,  tid);
  async_copy_lds(W2h,   ws + WS_W2,   INDIM * H2D, tid);
  async_copy_lds(W0dTh, ws + WS_W0DT, INDIM * H1D, tid);
  // per-block inputs (need f32->f16 convert anyway)
  for (int i = tid; i < 16 * INSZ; i += 256) {
    int s = i / INSZ, c = i - s * INSZ;
    float v = (c < INDIM) ? eps[(size_t)(s0 + s) * INDIM + c]
                          : xc[(size_t)(s0 + s) * CONDD + (c - INDIM)];
    inph[i] = (_Float16)v;
  }
  for (int i = tid; i < 16 * INDIM; i += 256) ldsDiag[i] = 0.0f;
  asm volatile("s_wait_asynccnt 0x0" ::: "memory");
  __syncthreads();

  const int mrow = lane & 15;
  const int hi8  = (lane & 16) ? 8 : 0;

  // ---- forward: h0 = inp @ W0^T + b0 ; a0/hp0 (M=16, K=96, N=256)
  for (int nt = wave; nt < 16; nt += 8) {
    v8f acc = {};
#pragma unroll
    for (int k = 0; k < 3; ++k) {
      v16h a = frag_a(inph, INSZ, mrow, k * 32, lane);
      v16h b = frag_b(W0h, INSZ, nt * 16 + mrow, k * 32, lane);
      acc = wmma16(a, b, acc);
    }
#pragma unroll
    for (int r = 0; r < 8; ++r) {
      int srow = r + hi8;
      int col  = nt * 16 + mrow;
      float av, hpv;
      lipswish_pair(acc[r] + b0[col], beta, av, hpv);
      a0h[srow * H1D + col]  = (_Float16)av;
      hp0h[srow * H1D + col] = (_Float16)hpv;
    }
  }
  __syncthreads();

  // ---- forward: h1 = a0 @ W1^T + b1 ; a1/hp1 (K=256, N=256)
  for (int nt = wave; nt < 16; nt += 8) {
    v8f acc = {};
#pragma unroll
    for (int k = 0; k < 8; ++k) {
      v16h a = frag_a(a0h, H1D, mrow, k * 32, lane);
      v16h b = frag_b(W1h, H1D, nt * 16 + mrow, k * 32, lane);
      acc = wmma16(a, b, acc);
    }
#pragma unroll
    for (int r = 0; r < 8; ++r) {
      int srow = r + hi8;
      int col  = nt * 16 + mrow;
      float av, hpv;
      lipswish_pair(acc[r] + b1[col], beta, av, hpv);
      a1h[srow * H2D + col]  = (_Float16)av;
      hp1h[srow * H2D + col] = (_Float16)hpv;
    }
  }
  __syncthreads();

  // ---- forward: g = a1 @ W2^T + b2 ; out1 = eps + g  (K=256, N=64)
  if (wave < 4) {
    int nt = wave;
    v8f acc = {};
#pragma unroll
    for (int k = 0; k < 8; ++k) {
      v16h a = frag_a(a1h, H2D, mrow, k * 32, lane);
      v16h b = frag_b(W2h, H2D, nt * 16 + mrow, k * 32, lane);
      acc = wmma16(a, b, acc);
    }
#pragma unroll
    for (int r = 0; r < 8; ++r) {
      int srow = r + hi8;
      int o    = nt * 16 + mrow;
      size_t gi = (size_t)(s0 + srow) * INDIM + o;
      out[gi] = eps[gi] + acc[r] + b2[o];
    }
  }

  // ---- stage 2: T = W1 @ (hp0 (*) W0d) per sample; contract with
  //      W2[o,h2]*hp1[h2] in VALU; reduce into ldsDiag via ds atomics.
  // Two samples per pass: shared A fragments (W1) -> half the LDS traffic.
  {
    const int nt2 = wave & 3;   // o-tile
    const int mh  = wave >> 2;  // half of the 16 h2-tiles
    const int o   = nt2 * 16 + mrow;
    for (int s = 0; s < 16; s += 2) {
      v16h bf0[8], bf1[8];
#pragma unroll
      for (int k = 0; k < 8; ++k) {
        v16h wv = frag_b(W0dTh, H1D, o, k * 32, lane);
        bf0[k] = wv * frag_b(hp0h, H1D, s,     k * 32, lane);
        bf1[k] = wv * frag_b(hp0h, H1D, s + 1, k * 32, lane);
      }
      float p0 = 0.0f, p1 = 0.0f;
      for (int mt = 0; mt < 8; ++mt) {
        int mtile = mh * 8 + mt;
        v8f acc0 = {}, acc1 = {};
#pragma unroll
        for (int k = 0; k < 8; ++k) {
          v16h a = frag_a(W1h, H1D, mtile * 16 + mrow, k * 32, lane);
          acc0 = wmma16(a, bf0[k], acc0);
          acc1 = wmma16(a, bf1[k], acc1);
        }
        int h2b = mtile * 16 + hi8;
        v8h w2v  = *(const v8h*)&W2h[o * H2D + h2b];
        v8h hpv0 = *(const v8h*)&hp1h[s * H2D + h2b];
        v8h hpv1 = *(const v8h*)&hp1h[(s + 1) * H2D + h2b];
#pragma unroll
        for (int r = 0; r < 8; ++r) {
          float wf = (float)w2v[r];
          p0 = fmaf(acc0[r] * wf, (float)hpv0[r], p0);
          p1 = fmaf(acc1[r] * wf, (float)hpv1[r], p1);
        }
      }
      p0 += __shfl_xor(p0, 16);
      p1 += __shfl_xor(p1, 16);
      if (lane < 16) {
        atomicAdd(&ldsDiag[s * INDIM + o], p0);
        atomicAdd(&ldsDiag[(s + 1) * INDIM + o], p1);
      }
    }
  }
  __syncthreads();

  // ---- j[s] = sum_o log|diag[s][o] + 1|
  {
    int s   = tid >> 4;
    int l16 = tid & 15;
    float v = 0.0f;
#pragma unroll
    for (int q = 0; q < 4; ++q) {
      float d = ldsDiag[s * INDIM + q * 16 + l16] + 1.0f;
      v += __logf(fabsf(d));
    }
    v += __shfl_xor(v, 8);
    v += __shfl_xor(v, 4);
    v += __shfl_xor(v, 2);
    v += __shfl_xor(v, 1);
    if (l16 == 0) out[(size_t)NSAMP * INDIM + s0 + s] = v;
  }
}

extern "C" void kernel_launch(void* const* d_in, const int* in_sizes, int n_in,
                              void* d_out, int out_size, void* d_ws,
                              size_t ws_size, hipStream_t stream) {
  const float* eps = (const float*)d_in[0];
  const float* x   = (const float*)d_in[1];
  const float* W0  = (const float*)d_in[2];
  const float* b0  = (const float*)d_in[3];
  const float* W1  = (const float*)d_in[4];
  const float* b1  = (const float*)d_in[5];
  const float* W2  = (const float*)d_in[6];
  const float* b2  = (const float*)d_in[7];
  const float* be  = (const float*)d_in[8];
  float* out = (float*)d_out;
  _Float16* ws = (_Float16*)d_ws;

  glrb_convert<<<dim3(64), dim3(256), 0, stream>>>(W0, W1, W2, ws);
  glrb_kernel<<<dim3(NSAMP / 16), dim3(256), 0, stream>>>(
      eps, x, ws, b0, b1, b2, be, out);
}
